// Mamba4CTRV3_75479755260593
// MI455X (gfx1250) — matrix operand order repaired and checked by reading, not compiled, not run on hardware
//
#include <hip/hip_runtime.h>
#include <hip/hip_bf16.h>
#include <math.h>

typedef __attribute__((ext_vector_type(16))) _Float16 v16h;
typedef __attribute__((ext_vector_type(8)))  float    v8f;

#define SEQL 40
#define RPAD 48          // rows padded to 3 x 16 WMMA tiles
#define DIM  64
#define NSP  26
#define NDD  13
#define VOC  10000
#define NLAY 4
#define HEADS 8
#define HD    8

// One 16x16 output tile via v_wmma_f32_16x16x32_f16, K unrolled in steps of 32.
// A: LDS f16 row-major [row][k], RPAD rows (pad rows zeroed -> no clamping).
// B: global f32 weight, row-major [N][K] (W as given; computes X @ W^T).
// CMODE: 0 = Cf[row][col] = acc + bias ; 1 = Ch[row][col] = f16(acc + bias) ;
//        2 = Cf[row][col] += acc + bias (fused residual add)
// All fragment loads/stores are unconditional (C buffers have RPAD rows).
template<int CMODE, int K, int NT>
__device__ __forceinline__ void gemm_tiles(
    const _Float16* __restrict__ A, int lda,
    const float* __restrict__ Wg,
    float* Cf, _Float16* Ch, int ldc,
    const float* __restrict__ bias, int tid)
{
  const int wave = tid >> 5;
  const int lane = tid & 31;
  const int r  = lane & 15;
  const int hi = lane >> 4;
  for (int tile = wave; tile < 3 * NT; tile += 8) {
    const int mt = tile / NT, nt = tile % NT;
    const int row0 = mt * 16, col0 = nt * 16;
    const _Float16* Ap = A + (row0 + r) * lda;
    const float*    Wp = Wg + (size_t)(col0 + r) * K;
    v8f acc = {};
#pragma unroll
    for (int k0 = 0; k0 < K; k0 += 32) {
      v16h a, b;
#pragma unroll
      for (int j = 0; j < 8; ++j) {               // ISA 16-bit A 16x32 layout
        a[j]     = Ap[k0 + hi * 8 + j];           // K = {0..7}+hi*8
        a[j + 8] = Ap[k0 + 16 + hi * 8 + j];      // K = 16+{0..7}+hi*8
      }
#pragma unroll
      for (int j = 0; j < 8; ++j) {               // B: lane holds col, K matches A layout
        b[j]     = (_Float16)Wp[k0 + hi * 8 + j];
        b[j + 8] = (_Float16)Wp[k0 + 16 + hi * 8 + j];
      }
      acc = __builtin_amdgcn_wmma_f32_16x16x32_f16(
              false, a, false, b, (short)0, acc, false, false);
    }
    const int col = col0 + r;
    const float bv = bias[col];
#pragma unroll
    for (int v = 0; v < 8; ++v) {                 // C/D layout: M = v + 8*hi, N = r
      const int row = row0 + v + 8 * hi;
      const float val = acc[v] + bv;
      if (CMODE == 0)      Cf[row * ldc + col]  = val;
      else if (CMODE == 1) Ch[row * ldc + col]  = (_Float16)val;
      else                 Cf[row * ldc + col] += val;
    }
  }
}

__global__ __launch_bounds__(256)
void mamba4ctr_fused(
    const float* __restrict__ dense_x,   const float* __restrict__ dense_W,
    const float* __restrict__ dense_b,   const float* __restrict__ embed_tables,
    const float* __restrict__ cls_token, const float* __restrict__ mb_norm_g,
    const float* __restrict__ mb_norm_b, const float* __restrict__ mb_xproj_W,
    const float* __restrict__ mb_xproj_b,const float* __restrict__ mb_A,
    const float* __restrict__ mb_Dp,     const float* __restrict__ mb_out_W,
    const float* __restrict__ mb_out_b,  const float* __restrict__ merge_W,
    const float* __restrict__ merge_b,   const float* __restrict__ attn_in_W,
    const float* __restrict__ attn_in_b, const float* __restrict__ attn_out_W,
    const float* __restrict__ attn_out_b,const float* __restrict__ mlp_W1,
    const float* __restrict__ mlp_b1,    const float* __restrict__ mlp_W2,
    const float* __restrict__ mlp_b2,    const float* __restrict__ mlp_W3,
    const float* __restrict__ mlp_b3,    const int* __restrict__ sparse_x,
    float* __restrict__ out)
{
  __shared__ float    s_seq[RPAD * DIM];          // 12 KB residual stream (C dest)
  __shared__ float    s_xn [RPAD * DIM];          // 12 KB LN out / oW GEMM dest
  __shared__ _Float16 s_xp [RPAD * 2 * DIM];      // 12 KB xproj out -> prod/gate (f16)
  __shared__ _Float16 s_abig[RPAD * 3 * DIM];     // 18 KB fr-concat (lda=128) / qkv (lda=192)
  __shared__ _Float16 s_ah [RPAD * DIM];          // 6 KB  f16 A-operand staging
  __shared__ float    s_mlp[192];

  const int b   = blockIdx.x;
  const int tid = threadIdx.x;

  // zero pad rows of the A-operand buffers (once; never overwritten for rows >= 40)
  for (int i = tid; i < (RPAD - SEQL) * DIM; i += 256)
    s_ah[SEQL * DIM + i] = (_Float16)0.f;
  for (int i = tid; i < (RPAD - SEQL) * 3 * DIM; i += 256)
    s_abig[SEQL * 3 * DIM + i] = (_Float16)0.f;

  // ---- build sequence: [cls | 13 dense | 26 sparse] ----
  for (int i = tid; i < DIM; i += 256) s_seq[i] = cls_token[i];
  for (int o = tid; o < NDD * DIM; o += 256) {
    float acc = dense_b[o];
    const float* w = dense_W + o * NDD;
    const float* x = dense_x + b * NDD;
#pragma unroll
    for (int j = 0; j < NDD; ++j) acc += x[j] * w[j];
    s_seq[DIM + o] = acc;
  }
  for (int i = tid; i < NSP * DIM; i += 256) {
    const int s = i / DIM, d = i % DIM;
    int idx = sparse_x[b * NSP + s];
    idx = idx < 0 ? 0 : (idx > VOC ? VOC : idx);
    s_seq[(1 + NDD + s) * DIM + d] =
        embed_tables[((size_t)s * (VOC + 1) + (size_t)idx) * DIM + d];
  }
  __syncthreads();

  // ---- 4 bidirectional Mamba layers ----
  for (int layer = 0; layer < NLAY; ++layer) {
    for (int dir = 0; dir < 2; ++dir) {
      const int wi = layer * 2 + dir;
      // LayerNorm, one row per thread
      if (tid < SEQL) {
        const float* xr = s_seq + tid * DIM;
        float m = 0.f, ss = 0.f;
#pragma unroll
        for (int d = 0; d < DIM; ++d) { float v = xr[d]; m += v; ss += v * v; }
        m *= (1.f / DIM);
        float inv = rsqrtf(ss * (1.f / DIM) - m * m + 1e-5f);
        const float* g  = mb_norm_g + wi * DIM;
        const float* be = mb_norm_b + wi * DIM;
#pragma unroll
        for (int d = 0; d < DIM; ++d) {
          float v = (xr[d] - m) * inv * g[d] + be[d];
          s_xn[tid * DIM + d] = v;
          s_ah[tid * DIM + d] = (_Float16)v;
        }
      }
      __syncthreads();
      // xp = xn @ xW^T + xb   (40x64 @ 64x128), stored f16
      gemm_tiles<1, DIM, 8>(s_ah, DIM, mb_xproj_W + (size_t)wi * 2 * DIM * DIM,
                            nullptr, s_xp, 2 * DIM, mb_xproj_b + wi * 2 * DIM, tid);
      __syncthreads();
      // softplus(delta), gate = sum_n exp(delta*A[d][n]); prod = xn * Bm
      for (int i = tid; i < SEQL * DIM; i += 256) {
        const int t = i / DIM, d = i % DIM;
        const float delta = (float)s_xp[t * 2 * DIM + d];
        const float Bm    = (float)s_xp[t * 2 * DIM + DIM + d];
        const float sp = fmaxf(delta, 0.f) + __logf(1.f + __expf(-fabsf(delta)));
        const float* Arow = mb_A + ((size_t)wi * DIM + d) * 16;
        float gsum = 0.f;
#pragma unroll
        for (int n = 0; n < 16; ++n) gsum += __expf(sp * Arow[n]);
        s_xp[t * 2 * DIM + d]       = (_Float16)(s_xn[i] * Bm);  // prod
        s_xp[t * 2 * DIM + DIM + d] = (_Float16)gsum;            // gate
      }
      __syncthreads();
      // scan (prefix fwd, suffix rev == reverse-scan-reverse), o = h*gate + xn*Dp
      if (tid < DIM) {
        const float Dpv = mb_Dp[wi * DIM + tid];
        float acc = 0.f;
        if (dir == 0) {
          for (int s = 0; s < SEQL; ++s) {
            acc += (float)s_xp[s * 2 * DIM + tid];
            float o = acc * (float)s_xp[s * 2 * DIM + DIM + tid]
                    + s_xn[s * DIM + tid] * Dpv;
            s_ah[s * DIM + tid] = (_Float16)o;
          }
        } else {
          for (int s = SEQL - 1; s >= 0; --s) {
            acc += (float)s_xp[s * 2 * DIM + tid];
            float o = acc * (float)s_xp[s * 2 * DIM + DIM + tid]
                    + s_xn[s * DIM + tid] * Dpv;
            s_ah[s * DIM + tid] = (_Float16)o;
          }
        }
      }
      __syncthreads();
      // o @ oW^T + ob  (40x64 @ 64x64) -> overwrite s_xn
      gemm_tiles<0, DIM, 4>(s_ah, DIM, mb_out_W + (size_t)wi * DIM * DIM,
                            s_xn, nullptr, DIM, mb_out_b + wi * DIM, tid);
      __syncthreads();
      // residual add, stage into concat buffer (f|r) as f16
      for (int i = tid; i < SEQL * DIM; i += 256) {
        const int t = i / DIM, d = i % DIM;
        s_abig[t * 2 * DIM + dir * DIM + d] = (_Float16)(s_xn[i] + s_seq[i]);
      }
      __syncthreads();
    }
    // merge: (40x128) @ (128x64) -> new seq
    gemm_tiles<0, 2 * DIM, 4>(s_abig, 2 * DIM, merge_W + (size_t)layer * DIM * 2 * DIM,
                              s_seq, nullptr, DIM, merge_b + layer * DIM, tid);
    __syncthreads();
  }

  // ---- attention ----
  for (int i = tid; i < SEQL * DIM; i += 256) s_ah[i] = (_Float16)s_seq[i];
  __syncthreads();
  // qkv = seq @ attn_in_W^T + b  (40x64 @ 64x192), stored f16 (pad rows get bias: finite)
  gemm_tiles<1, DIM, 12>(s_ah, DIM, attn_in_W, nullptr, s_abig, 3 * DIM, attn_in_b, tid);
  __syncthreads();
  // per (head, query) softmax attention, head dim 8 on VALU
  for (int task = tid; task < HEADS * SEQL; task += 256) {
    const int h = task / SEQL, q = task % SEQL;
    float sc[SEQL];
    float mx = -1e30f;
    const _Float16* qp = s_abig + q * 3 * DIM + h * HD;
    for (int k = 0; k < SEQL; ++k) {
      const _Float16* kp = s_abig + k * 3 * DIM + DIM + h * HD;
      float s = 0.f;
#pragma unroll
      for (int d = 0; d < HD; ++d) s += (float)qp[d] * (float)kp[d];
      s *= 0.35355339059327f;  // 1/sqrt(8)
      sc[k] = s;
      mx = fmaxf(mx, s);
    }
    float den = 0.f;
    for (int k = 0; k < SEQL; ++k) { float e = __expf(sc[k] - mx); sc[k] = e; den += e; }
    const float dinv = 1.f / den;
#pragma unroll
    for (int d = 0; d < HD; ++d) {
      float ao = 0.f;
      for (int k = 0; k < SEQL; ++k)
        ao += sc[k] * (float)s_abig[k * 3 * DIM + 2 * DIM + h * HD + d];
      s_xn[q * DIM + h * HD + d] = ao * dinv;
    }
  }
  __syncthreads();
  for (int i = tid; i < SEQL * DIM; i += 256) s_ah[i] = (_Float16)s_xn[i];
  __syncthreads();
  // seq += ao @ attn_out_W^T + b  (fused residual via CMODE 2)
  gemm_tiles<2, DIM, 4>(s_ah, DIM, attn_out_W, s_seq, nullptr, DIM, attn_out_b, tid);
  __syncthreads();

  // ---- MLP on cls row ----
  if (tid < 128) {
    float acc = mlp_b1[tid];
    const float* w = mlp_W1 + tid * DIM;
#pragma unroll
    for (int k = 0; k < DIM; ++k) acc += s_seq[k] * w[k];
    s_mlp[tid] = fmaxf(acc, 0.f);
  }
  __syncthreads();
  if (tid < 64) {
    float acc = mlp_b2[tid];
    const float* w = mlp_W2 + tid * 128;
#pragma unroll
    for (int k = 0; k < 128; ++k) acc += s_mlp[k] * w[k];
    s_mlp[128 + tid] = fmaxf(acc, 0.f);
  }
  __syncthreads();
  if (tid == 0) {
    float acc = mlp_b3[0];
#pragma unroll
    for (int k = 0; k < 64; ++k) acc += s_mlp[128 + k] * mlp_W3[k];
    out[b] = acc;
  }
}

extern "C" void kernel_launch(void* const* d_in, const int* in_sizes, int n_in,
                              void* d_out, int out_size, void* d_ws, size_t ws_size,
                              hipStream_t stream) {
  (void)n_in; (void)d_ws; (void)ws_size; (void)out_size;
  const float* dense_x      = (const float*)d_in[0];
  const float* dense_W      = (const float*)d_in[1];
  const float* dense_b      = (const float*)d_in[2];
  const float* embed_tables = (const float*)d_in[3];
  const float* cls_token    = (const float*)d_in[4];
  const float* mb_norm_g    = (const float*)d_in[5];
  const float* mb_norm_b    = (const float*)d_in[6];
  const float* mb_xproj_W   = (const float*)d_in[7];
  const float* mb_xproj_b   = (const float*)d_in[8];
  const float* mb_A         = (const float*)d_in[9];
  const float* mb_Dv        = (const float*)d_in[10];
  const float* mb_out_W     = (const float*)d_in[11];
  const float* mb_out_b     = (const float*)d_in[12];
  const float* merge_W      = (const float*)d_in[13];
  const float* merge_b      = (const float*)d_in[14];
  const float* attn_in_W    = (const float*)d_in[15];
  const float* attn_in_b    = (const float*)d_in[16];
  const float* attn_out_W   = (const float*)d_in[17];
  const float* attn_out_b   = (const float*)d_in[18];
  const float* mlp_W1       = (const float*)d_in[19];
  const float* mlp_b1       = (const float*)d_in[20];
  const float* mlp_W2       = (const float*)d_in[21];
  const float* mlp_b2       = (const float*)d_in[22];
  const float* mlp_W3       = (const float*)d_in[23];
  const float* mlp_b3       = (const float*)d_in[24];
  const int*   sparse_x     = (const int*)d_in[25];

  const int Bsz = in_sizes[0] / NDD;   // dense_x is (B, 13)
  mamba4ctr_fused<<<Bsz, 256, 0, stream>>>(
      dense_x, dense_W, dense_b, embed_tables, cls_token,
      mb_norm_g, mb_norm_b, mb_xproj_W, mb_xproj_b, mb_A, mb_Dv,
      mb_out_W, mb_out_b, merge_W, merge_b,
      attn_in_W, attn_in_b, attn_out_W, attn_out_b,
      mlp_W1, mlp_b1, mlp_W2, mlp_b2, mlp_W3, mlp_b3,
      sparse_x, (float*)d_out);
}
